// MultiheadAttention_42872363548737
// MI455X (gfx1250) — matrix operand order repaired and checked by reading, not compiled, not run on hardware
//
#include <hip/hip_runtime.h>
#include <stdint.h>

// ---------------- problem constants (Enformer/Borzoi MHA) ----------------
constexpr int Bb = 2;            // batch
constexpr int Hh = 8;            // heads
constexpr int Tt = 1536;         // sequence
constexpr int Cc = 1536;         // input channels
constexpr int Kk = 64;           // key size
constexpr int Vv = 192;          // value size
constexpr int HK = Hh * Kk;      // 512
constexpr int HV = Hh * Vv;      // 1536
constexpr int RKROWS = 2 * Tt + 16;   // 2T-1 rows + zero padding for band GEMM

// ---------------- WMMA types / helpers ----------------
typedef __attribute__((ext_vector_type(16))) __bf16 v16bf;
typedef __attribute__((ext_vector_type(8)))  float  v8f;
typedef __attribute__((ext_vector_type(4)))  int    v4i;

struct __align__(16) U128 { unsigned int x, y, z, w; };
struct __align__(8)  U64  { unsigned int x, y; };
union BF16x16 { v16bf v; U128 u[2]; };
union BF16x4  { __bf16 b[4]; U64 u; };

// A operand (16x32 bf16, M rows x K): lane<16 holds k[0..7],[16..23] of row lm;
// lane>=16 holds k[8..15],[24..31] of row lm.
__device__ __forceinline__ v16bf load_a16(const __bf16* row0, int ld, int lane) {
  const int half = (lane >> 4) & 1;
  const int lm   = lane & 15;
  const __bf16* p = row0 + lm * ld + 8 * half;
  BF16x16 r;
  r.u[0] = *(const U128*)(p);
  r.u[1] = *(const U128*)(p + 16);
  return r.v;
}

// B operand (32x16 bf16, K x N): lane<16 holds kk[0..7],[16..23] of col lm; lane>=16 kk[8..15],[24..31].
__device__ __forceinline__ v16bf load_b16(const __bf16* p0, int ld, int lane) {
  const int half = (lane >> 4) & 1;
  const int lm   = lane & 15;
  const __bf16* p = p0 + lm * ld + 8 * half;
  BF16x16 r;
  r.u[0] = *(const U128*)(p);
  r.u[1] = *(const U128*)(p + 16);
  return r.v;
}

__device__ __forceinline__ v8f wmma_bf16(v16bf a, v16bf b, v8f c) {
  return __builtin_amdgcn_wmma_f32_16x16x32_bf16(false, a, false, b, (short)0, c,
                                                 false, false);
}

// 16-byte global -> LDS copy via CDNA5 async DMA (ASYNCcnt-tracked, no VGPR
// staging); inline-asm / synchronous fallbacks kept for other toolchains.
__device__ __forceinline__ void async_cp16(const __bf16* g, __bf16* l) {
#if __has_builtin(__builtin_amdgcn_global_load_async_to_lds_b128)
  __builtin_amdgcn_global_load_async_to_lds_b128(
      (__attribute__((address_space(1))) v4i*)(void*)g,
      (__attribute__((address_space(3))) v4i*)(void*)l, 0, 0);
#elif defined(__gfx1250__)
  unsigned lofs = (unsigned)(unsigned long long)
      (__attribute__((address_space(3))) __bf16*)l;
  asm volatile("global_load_async_to_lds_b128 %0, %1, off"
               :: "v"(lofs), "v"((unsigned long long)g) : "memory");
#else
  *(U128*)l = *(const U128*)g;
#endif
}

__device__ __forceinline__ void wait_async0() {
#if __has_builtin(__builtin_amdgcn_s_wait_asynccnt)
  __builtin_amdgcn_s_wait_asynccnt(0);
#elif defined(__gfx1250__)
  asm volatile("s_wait_asynccnt 0" ::: "memory");
#endif
}

// ---------------- generic tiled WMMA GEMM: C[M,N] = A[M,Kd] * B[Kd,N] (+bias) ----
// fp32 in / fp32 out, bf16 compute. Block: 256 thr (8 waves), tile 128x64, BK=32,
// double-buffered LDS so global loads for tile i+1 overlap the WMMAs of tile i.
template <int Kd, int N>
__global__ __launch_bounds__(256) void gemm_bf16_kernel(
    const float* __restrict__ A, const float* __restrict__ Bw,
    float* __restrict__ C, const float* __restrict__ bias)
{
  __shared__ __align__(16) __bf16 As[2][128][32];
  __shared__ __align__(16) __bf16 Bs[2][64][32];   // stored transposed: [n][kk]

  const int m0 = blockIdx.y * 128, n0 = blockIdx.x * 64;
  const int tid = threadIdx.x;
  const int wave = tid >> 5, lane = tid & 31;
  const int rw = (wave >> 1) * 32, cw = (wave & 1) * 32;

  const v8f z = {0.f, 0.f, 0.f, 0.f, 0.f, 0.f, 0.f, 0.f};
  v8f acc[2][2] = {{z, z}, {z, z}};

  const int ac4 = tid & 7,  ar = tid >> 3;   // A: 8 float4-cols x 32 rows
  const int bc4 = tid & 15, br = tid >> 4;   // B: 16 float4-cols x 16 rows

  float4 fa[4], fb[2];
  auto gload = [&](int k0) {
#pragma unroll
    for (int i = 0; i < 4; ++i)
      fa[i] = *(const float4*)(A + (unsigned)((m0 + ar + 32 * i) * Kd + k0 + 4 * ac4));
#pragma unroll
    for (int i = 0; i < 2; ++i)
      fb[i] = *(const float4*)(Bw + (unsigned)((k0 + br + 16 * i) * N + n0 + 4 * bc4));
  };
  auto lstore = [&](int buf) {
#pragma unroll
    for (int i = 0; i < 4; ++i) {
      BF16x4 cv;
      cv.b[0] = (__bf16)fa[i].x; cv.b[1] = (__bf16)fa[i].y;
      cv.b[2] = (__bf16)fa[i].z; cv.b[3] = (__bf16)fa[i].w;
      *(U64*)(&As[buf][ar + 32 * i][4 * ac4]) = cv.u;
    }
#pragma unroll
    for (int i = 0; i < 2; ++i) {
      Bs[buf][4 * bc4 + 0][br + 16 * i] = (__bf16)fb[i].x;
      Bs[buf][4 * bc4 + 1][br + 16 * i] = (__bf16)fb[i].y;
      Bs[buf][4 * bc4 + 2][br + 16 * i] = (__bf16)fb[i].z;
      Bs[buf][4 * bc4 + 3][br + 16 * i] = (__bf16)fb[i].w;
    }
  };

  gload(0);
  lstore(0);
  int cur = 0;
  for (int k0 = 0; k0 < Kd; k0 += 32) {
    __syncthreads();                       // buf[cur] visible to all waves
    const bool more = (k0 + 32) < Kd;
    if (more) gload(k0 + 32);              // overlap with WMMAs below

    v16bf a0 = load_a16(&As[cur][rw][0],      32, lane);
    v16bf a1 = load_a16(&As[cur][rw + 16][0], 32, lane);
    v16bf b0 = load_b16(&Bs[cur][cw][0],      32, lane);
    v16bf b1 = load_b16(&Bs[cur][cw + 16][0], 32, lane);
    acc[0][0] = wmma_bf16(a0, b0, acc[0][0]);
    acc[0][1] = wmma_bf16(a0, b1, acc[0][1]);
    acc[1][0] = wmma_bf16(a1, b0, acc[1][0]);
    acc[1][1] = wmma_bf16(a1, b1, acc[1][1]);

    if (more) lstore(cur ^ 1);
    cur ^= 1;
  }

  const int half = lane >> 4, lm = lane & 15;
  float bv0 = 0.f, bv1 = 0.f;
  if (bias != nullptr) {                   // uniform branch, 2 loads total
    bv0 = bias[n0 + cw + lm];
    bv1 = bias[n0 + cw + 16 + lm];
  }
#pragma unroll
  for (int rs = 0; rs < 2; ++rs)
#pragma unroll
    for (int cs = 0; cs < 2; ++cs)
#pragma unroll
      for (int g = 0; g < 8; ++g) {
        int row = m0 + rw + 16 * rs + g + 8 * half;
        int col = n0 + cw + 16 * cs + lm;
        C[(unsigned)(row * N + col)] = acc[rs][cs][g] + (cs ? bv1 : bv0);
      }
}

// ---------------- rk = positional_features(dist) @ Wr, packed bf16 [h][r][K] ----
__global__ __launch_bounds__(256) void rk_build_kernel(
    const float* __restrict__ Wr, __bf16* __restrict__ rkb)
{
  int idx = blockIdx.x * 256 + threadIdx.x;     // [0, RKROWS*512)
  int col = idx & 511;                          // h*64 + k
  int r   = idx >> 9;
  int h = col >> 6, k = col & 63;
  float acc = 0.f;
  if (r < 2 * Tt - 1) {
    float d = (float)(r - (Tt - 1));
    float s = (d > 0.f) ? 1.f : ((d < 0.f) ? -1.f : 0.f);
    float a = fabsf(d);
    const float l2pr = log2f((float)(Tt + 1)) * (1.f / 16.f);
#pragma unroll
    for (int i = 0; i < 16; ++i) {
      float cw = exp2f((float)(i + 1) * l2pr) - 1.f;   // center widths
      if (cw > a) acc += Wr[i * 512 + col] + s * Wr[(16 + i) * 512 + col];
    }
  }
  rkb[(unsigned)((h * RKROWS + r) * 64 + k)] = (__bf16)acc;
}

// ---------------- pack q (scaled + both biases) and k into bf16 [b,h,t,K] ------
__global__ __launch_bounds__(256) void pack_qk_kernel(
    const float* __restrict__ qf, const float* __restrict__ kf,
    const float* __restrict__ rwb, const float* __restrict__ rrb,
    __bf16* __restrict__ qc, __bf16* __restrict__ qr, __bf16* __restrict__ kp)
{
  int idx = blockIdx.x * 256 + threadIdx.x;     // [0, B*T*512)
  int col = idx & 511;                          // h*64+k
  int bt  = idx >> 9;
  int h = col >> 6, k = col & 63;
  int b = bt / Tt, t = bt % Tt;
  float qv = qf[idx] * 0.125f;                  // K^-0.5, K=64
  unsigned dst = (unsigned)((((b * Hh + h)) * Tt + t) * 64 + k);
  qc[dst] = (__bf16)(qv + rwb[col]);
  qr[dst] = (__bf16)(qv + rrb[col]);
  kp[dst] = (__bf16)kf[idx];
}

// ---------------- v: [b,t,h,v] f32 -> [b,h,v,t] bf16 (LDS tile transpose) -------
__global__ __launch_bounds__(256) void transpose_v_kernel(
    const float* __restrict__ vf, __bf16* __restrict__ vT)
{
  __shared__ float tile[32][33];
  int bx = blockIdx.x;
  int nb = bx % (Vv / 32); bx /= (Vv / 32);
  int tb = bx % (Tt / 32); bx /= (Tt / 32);
  int h  = bx % Hh;        int b = bx / Hh;
  int c = threadIdx.x & 31, r4 = threadIdx.x >> 5;
#pragma unroll
  for (int i = 0; i < 4; ++i) {
    int tl = r4 + 8 * i;
    tile[tl][c] = vf[(unsigned)((b * Tt + tb * 32 + tl) * HV + h * Vv + nb * 32 + c)];
  }
  __syncthreads();
#pragma unroll
  for (int i = 0; i < 4; ++i) {
    int nl = r4 + 8 * i;
    vT[(unsigned)(((b * Hh + h) * Vv + nb * 32 + nl) * Tt + tb * 32 + c)] =
        (__bf16)tile[c][nl];
  }
}

// ---------------- fused attention: content + shifted rel + softmax + P*V --------
// grid: B*H*(T/64) blocks, 128 threads (4 waves); each wave owns 16 query rows.
// Per 32-wide s-step, k/rk-band/v tiles are staged to LDS with async-to-LDS DMA,
// double-buffered so tile i+1's fill overlaps tile i's 22 WMMAs.
__global__ __launch_bounds__(128) void attn_kernel(
    const __bf16* __restrict__ qc, const __bf16* __restrict__ qr,
    const __bf16* __restrict__ kp, const __bf16* __restrict__ vT,
    const __bf16* __restrict__ rkb, float* __restrict__ outp)
{
  const int ntb = Tt / 64;
  const int bh = blockIdx.x / ntb;
  const int tb = blockIdx.x % ntb;
  const int b = bh >> 3, h = bh & 7;
  const int tid = threadIdx.x;
  const int wave = tid >> 5, lane = tid & 31;
  const int half = lane >> 4, lm = lane & 15;
  const int tw0 = tb * 64 + wave * 16;

  const __bf16* qc_p = qc + ((size_t)bh * Tt + tw0) * 64;
  const __bf16* qr_p = qr + ((size_t)bh * Tt + tw0) * 64;
  const __bf16* kp_p = kp + (size_t)bh * Tt * 64;
  const __bf16* vT_p = vT + (size_t)bh * Vv * Tt;
  const __bf16* rk_p = rkb + (size_t)h * RKROWS * 64;

  // double-buffered LDS tiles (28KB/buffer) + per-wave P staging (4KB)
  __shared__ __align__(16) __bf16 kS [2][32][64];    // k rows s0..s0+31
  __shared__ __align__(16) __bf16 rkS[2][96][64];    // rk band union of 4 waves
  __shared__ __align__(16) __bf16 vS [2][192][32];   // v cols, 32 s-entries each
  __shared__ __align__(16) __bf16 plds[4][16][32];
  __bf16* pw = &plds[wave][0][0];

  // cooperative async staging of one s-step into buffer `buf`
  auto stage = [&](int buf, int s0n) {
    // k tile: 4KB contiguous -> 2 x b128 per thread
    const __bf16* ksrc = kp_p + (size_t)s0n * 64;
#pragma unroll
    for (int i = 0; i < 2; ++i) {
      int c = tid + 128 * i;
      async_cp16(ksrc + c * 8, &kS[buf][0][0] + c * 8);
    }
    // rk band: rows S0..S0+95 (12KB contiguous) -> 6 x b128 per thread
    const int S0 = (Tt - 1) + s0n - tb * 64 - 63;    // wave0 base - 48, >= 0
    const __bf16* rsrc = rk_p + (size_t)S0 * 64;
#pragma unroll
    for (int i = 0; i < 6; ++i) {
      int c = tid + 128 * i;
      async_cp16(rsrc + c * 8, &rkS[buf][0][0] + c * 8);
    }
    // v: 192 rows x 64B segments from vT rows -> 6 x b128 per thread
#pragma unroll
    for (int i = 0; i < 6; ++i) {
      int c = tid + 128 * i;          // 768 16B chunks
      int row = c >> 2, part = c & 3;
      async_cp16(vT_p + (size_t)row * Tt + s0n + part * 8,
                 &vS[buf][0][0] + row * 32 + part * 8);
    }
  };

  // persistent query operands (content and rel variants), K=64 -> 2 chunks
  v16bf qa0 = load_a16(qc_p,      64, lane);
  v16bf qa1 = load_a16(qc_p + 32, 64, lane);
  v16bf qb0 = load_a16(qr_p,      64, lane);
  v16bf qb1 = load_a16(qr_p + 32, 64, lane);

  const v8f z = {0.f, 0.f, 0.f, 0.f, 0.f, 0.f, 0.f, 0.f};
  v8f acc[12];
#pragma unroll
  for (int i = 0; i < 12; ++i) acc[i] = z;
  float mrow[8], lrow[8];
#pragma unroll
  for (int g = 0; g < 8; ++g) { mrow[g] = -__builtin_inff(); lrow[g] = 0.f; }

  stage(0, 0);
  int cur = 0;
  for (int s0 = 0; s0 < Tt; s0 += 32) {
    wait_async0();                 // this wave's staged DMA done
    __syncthreads();               // all waves' tiles visible
    if (s0 + 32 < Tt) stage(cur ^ 1, s0 + 32);   // overlap with compute below

    const __bf16* kS_ = &kS[cur][0][0];
    const __bf16* rS_ = &rkS[cur][48 - 16 * wave][0];  // this wave's band start
    const __bf16* vS_ = &vS[cur][0][0];

    // content logits 16x32 (2 col-subtiles x 2 k-chunks) from LDS
    v8f lg0 = wmma_bf16(qa0, load_b16(kS_,               64, lane), z);
    lg0     = wmma_bf16(qa1, load_b16(kS_ + 32,          64, lane), lg0);
    v8f lg1 = wmma_bf16(qa0, load_b16(kS_ + 16 * 64,     64, lane), z);
    lg1     = wmma_bf16(qa1, load_b16(kS_ + 16 * 64 + 32, 64, lane), lg1);

    // relative band: Drel[dt,j] = qr . rk[base+j], j in [0,48)
    v8f dr0 = wmma_bf16(qb0, load_b16(rS_,                64, lane), z);
    dr0     = wmma_bf16(qb1, load_b16(rS_ + 32,           64, lane), dr0);
    v8f dr1 = wmma_bf16(qb0, load_b16(rS_ + 16 * 64,      64, lane), z);
    dr1     = wmma_bf16(qb1, load_b16(rS_ + 16 * 64 + 32, 64, lane), dr1);

    // relative-shift gather into lg0 (dr0 dies here)
#pragma unroll
    for (int g = 0; g < 8; ++g) {
      int dt = g + 8 * half;
      int src = ((lm - dt + 15) & 15) + 16 * half;
      bool hi = (lm >= dt + 1);
      float l0 = __shfl(dr0[g], src, 32);
      float h0 = __shfl(dr1[g], src, 32);
      lg0[g] += hi ? h0 : l0;
    }

    v8f dr2 = wmma_bf16(qb0, load_b16(rS_ + 32 * 64,      64, lane), z);
    dr2     = wmma_bf16(qb1, load_b16(rS_ + 32 * 64 + 32, 64, lane), dr2);

#pragma unroll
    for (int g = 0; g < 8; ++g) {
      int dt = g + 8 * half;
      int src = ((lm - dt + 15) & 15) + 16 * half;
      bool hi = (lm >= dt + 1);
      float l1 = __shfl(dr1[g], src, 32);
      float h1 = __shfl(dr2[g], src, 32);
      lg1[g] += hi ? h1 : l1;
    }

    // online softmax (rows live in 16-lane halves; xor-reduce within half)
    float scl[8];
#pragma unroll
    for (int g = 0; g < 8; ++g) {
      float mt = fmaxf(lg0[g], lg1[g]);
#pragma unroll
      for (int d = 1; d < 16; d <<= 1) mt = fmaxf(mt, __shfl_xor(mt, d, 32));
      float mn = fmaxf(mrow[g], mt);
      scl[g] = __expf(mrow[g] - mn);
      mrow[g] = mn;
      lg0[g] = __expf(lg0[g] - mn);
      lg1[g] = __expf(lg1[g] - mn);
      float rs = lg0[g] + lg1[g];
#pragma unroll
      for (int d = 1; d < 16; d <<= 1) rs += __shfl_xor(rs, d, 32);
      lrow[g] = lrow[g] * scl[g] + rs;
    }
#pragma unroll
    for (int vt = 0; vt < 12; ++vt)
#pragma unroll
      for (int g = 0; g < 8; ++g) acc[vt][g] *= scl[g];

    // P: C-layout -> A-layout via per-wave LDS (same-wave LDS is in-order)
#pragma unroll
    for (int g = 0; g < 8; ++g) {
      int row = g + 8 * half;
      pw[row * 32 + lm]      = (__bf16)lg0[g];
      pw[row * 32 + 16 + lm] = (__bf16)lg1[g];
    }
    asm volatile("s_wait_dscnt 0" ::: "memory");
    v16bf pa = load_a16(pw, 32, lane);
    __builtin_amdgcn_wave_barrier();

    // acc += P(16x32) @ V(32x192): 12 WMMAs, B operands from LDS
#pragma unroll
    for (int vt = 0; vt < 12; ++vt)
      acc[vt] = wmma_bf16(pa, load_b16(vS_ + 16 * vt * 32, 32, lane), acc[vt]);

    cur ^= 1;
  }

  float rinv[8];
#pragma unroll
  for (int g = 0; g < 8; ++g) rinv[g] = 1.f / lrow[g];
#pragma unroll
  for (int vt = 0; vt < 12; ++vt)
#pragma unroll
    for (int g = 0; g < 8; ++g) {
      int row = tw0 + g + 8 * half;
      int col = h * Vv + 16 * vt + lm;
      outp[(unsigned)((b * Tt + row) * HV + col)] = acc[vt][g] * rinv[g];
    }
}

// ---------------- host-side orchestration ----------------
extern "C" void kernel_launch(void* const* d_in, const int* in_sizes, int n_in,
                              void* d_out, int out_size, void* d_ws, size_t ws_size,
                              hipStream_t stream) {
  (void)in_sizes; (void)n_in; (void)out_size; (void)ws_size;
  const float* x   = (const float*)d_in[0];
  const float* Wq  = (const float*)d_in[1];
  const float* Wk  = (const float*)d_in[2];
  const float* Wv  = (const float*)d_in[3];
  const float* Wr  = (const float*)d_in[4];
  const float* rwb = (const float*)d_in[5];
  const float* rrb = (const float*)d_in[6];
  const float* Wo  = (const float*)d_in[7];
  const float* bo  = (const float*)d_in[8];
  float* out = (float*)d_out;

  size_t off = 0;
  auto take = [&](size_t bytes) -> void* {
    void* p = (char*)d_ws + off;
    off += (bytes + 255) & ~(size_t)255;
    return p;
  };
  float*  qf    = (float*) take((size_t)Bb * Tt * HK * 4);
  float*  kf    = (float*) take((size_t)Bb * Tt * HK * 4);
  float*  vf    = (float*) take((size_t)Bb * Tt * HV * 4);
  float*  attnf = (float*) take((size_t)Bb * Tt * HV * 4);
  __bf16* qcb   = (__bf16*)take((size_t)Bb * Hh * Tt * Kk * 2);
  __bf16* qrb   = (__bf16*)take((size_t)Bb * Hh * Tt * Kk * 2);
  __bf16* kpb   = (__bf16*)take((size_t)Bb * Hh * Tt * Kk * 2);
  __bf16* vTb   = (__bf16*)take((size_t)Bb * Hh * Vv * Tt * 2);
  __bf16* rkbb  = (__bf16*)take((size_t)Hh * RKROWS * Kk * 2);

  const int mt = (Bb * Tt) / 128;   // 24 row tiles

  gemm_bf16_kernel<Cc, HK><<<dim3(HK / 64, mt), 256, 0, stream>>>(x, Wq, qf, nullptr);
  gemm_bf16_kernel<Cc, HK><<<dim3(HK / 64, mt), 256, 0, stream>>>(x, Wk, kf, nullptr);
  gemm_bf16_kernel<Cc, HV><<<dim3(HV / 64, mt), 256, 0, stream>>>(x, Wv, vf, nullptr);

  rk_build_kernel<<<(RKROWS * HK) / 256, 256, 0, stream>>>(Wr, rkbb);
  pack_qk_kernel<<<(Bb * Tt * HK) / 256, 256, 0, stream>>>(qf, kf, rwb, rrb,
                                                           qcb, qrb, kpb);
  transpose_v_kernel<<<Bb * Hh * (Tt / 32) * (Vv / 32), 256, 0, stream>>>(vf, vTb);

  attn_kernel<<<Bb * Hh * (Tt / 64), 128, 0, stream>>>(qcb, qrb, kpb, vTb, rkbb,
                                                       attnf);

  gemm_bf16_kernel<HV, HV><<<dim3(HV / 64, mt), 256, 0, stream>>>(attnf, Wo, out, bo);
}
